// Attn_61418032333261
// MI455X (gfx1250) — compile-verified, matrix operand order checked by reading, not run on hardware
//
#include <hip/hip_runtime.h>
#include <hip/hip_bf16.h>

// ---------- types ----------
typedef __bf16  bf16x16 __attribute__((ext_vector_type(16)));
typedef float   f32x8   __attribute__((ext_vector_type(8)));
typedef unsigned int u32x4 __attribute__((ext_vector_type(4)));
typedef int     i32x8   __attribute__((ext_vector_type(8)));
typedef int     i32x4   __attribute__((ext_vector_type(4)));

union Frag { bf16x16 v; u32x4 u[2]; };

__device__ __forceinline__ unsigned short f2bf(float f) {
    union { float f; unsigned u; } v; v.f = f;
    unsigned u = v.u;
    u += 0x7FFFu + ((u >> 16) & 1u);        // round-to-nearest-even
    return (unsigned short)(u >> 16);
}

__device__ __forceinline__ f32x8 wmma_bf16(bf16x16 a, bf16x16 b, f32x8 c) {
    return __builtin_amdgcn_wmma_f32_16x16x32_bf16(
        /*neg_a=*/false, a, /*neg_b=*/false, b,
        /*c_mod=*/(short)0, c, /*reuse_a=*/false, /*reuse_b=*/false);
}

// ---------- constants ----------
#define Dm   1024
#define Hh   16
#define Tt   2048
#define DHd  64
#define BT   4096          // B*T

// =====================================================================
// Elementwise f32 -> bf16 (4 elements per thread)
// =====================================================================
__global__ __launch_bounds__(256) void convert_bf16_kernel(
    const float* __restrict__ in, unsigned short* __restrict__ out)
{
    int gid = blockIdx.x * 256 + threadIdx.x;
    float4 f = *(const float4*)(in + (size_t)gid * 4);
    unsigned lo = (unsigned)f2bf(f.x) | ((unsigned)f2bf(f.y) << 16);
    unsigned hi = (unsigned)f2bf(f.z) | ((unsigned)f2bf(f.w) << 16);
    ((unsigned*)out)[(size_t)gid * 2]     = lo;
    ((unsigned*)out)[(size_t)gid * 2 + 1] = hi;
}

// =====================================================================
// Weight transpose + convert: W[K=1024][N=1024] f32 -> Wt[N][K] bf16
// =====================================================================
__global__ __launch_bounds__(256) void transpose_w_kernel(
    const float* __restrict__ W, unsigned short* __restrict__ Wt)
{
    __shared__ unsigned short s[32][33];
    const int kb = blockIdx.y * 32, nb = blockIdx.x * 32;
    const int c = threadIdx.x & 31, r0 = threadIdx.x >> 5;   // 8 row-groups
    for (int i = 0; i < 4; ++i) {
        int r = r0 + i * 8;
        s[r][c] = f2bf(W[(size_t)(kb + r) * Dm + nb + c]);
    }
    __syncthreads();
    for (int i = 0; i < 4; ++i) {
        int r = r0 + i * 8;
        Wt[(size_t)(nb + r) * Dm + kb + c] = s[c][r];
    }
}

// =====================================================================
// GEMM (bf16 operands, f32 accumulate):
//   out[M,N] = Ab[M,K] @ Wt[N,K]^T + bias[N]
// Fragments load straight from global (contiguous 16B runs per lane);
// operands are L2-resident so no LDS tiling is needed.
// mode 0: write f32 [M,N];  mode 1: write bf16 per-head-transposed V^T
// =====================================================================
__global__ __launch_bounds__(128) void gemm_bf16_kernel(
    const unsigned short* __restrict__ Ab,   // [M,K] bf16
    const unsigned short* __restrict__ Wt,   // [N,K] bf16
    const float* __restrict__ bias, void* __restrict__ out,
    int M, int N, int K, int mode)
{
    const int lane = threadIdx.x & 31;
    const int wave = threadIdx.x >> 5;
    const int bm = blockIdx.y * 64;
    const int bn = blockIdx.x * 64;
    const int wr = (wave >> 1) * 32;
    const int wc = (wave & 1) * 32;
    const int l16   = lane & 15;
    const int half8 = (lane >> 4) * 8;
    const int halfo = (lane >> 4) * 8;

    const unsigned short* a0p = Ab + (size_t)(bm + wr + l16) * K;
    const unsigned short* a1p = a0p + (size_t)16 * K;
    const unsigned short* b0p = Wt + (size_t)(bn + wc + l16) * K;
    const unsigned short* b1p = b0p + (size_t)16 * K;

    f32x8 acc[2][2] = {};
#pragma unroll 2
    for (int k0 = 0; k0 < K; k0 += 32) {
        Frag a0, a1, b0, b1;
        a0.u[0] = *(const u32x4*)(a0p + k0 + half8);
        a0.u[1] = *(const u32x4*)(a0p + k0 + half8 + 16);
        a1.u[0] = *(const u32x4*)(a1p + k0 + half8);
        a1.u[1] = *(const u32x4*)(a1p + k0 + half8 + 16);
        b0.u[0] = *(const u32x4*)(b0p + k0 + half8);
        b0.u[1] = *(const u32x4*)(b0p + k0 + half8 + 16);
        b1.u[0] = *(const u32x4*)(b1p + k0 + half8);
        b1.u[1] = *(const u32x4*)(b1p + k0 + half8 + 16);
        acc[0][0] = wmma_bf16(a0.v, b0.v, acc[0][0]);
        acc[0][1] = wmma_bf16(a0.v, b1.v, acc[0][1]);
        acc[1][0] = wmma_bf16(a1.v, b0.v, acc[1][0]);
        acc[1][1] = wmma_bf16(a1.v, b1.v, acc[1][1]);
    }

    for (int mi = 0; mi < 2; ++mi)
        for (int ni = 0; ni < 2; ++ni)
            for (int i = 0; i < 8; ++i) {
                int row = bm + wr + mi * 16 + halfo + i;
                int col = bn + wc + ni * 16 + l16;
                float v = acc[mi][ni][i] + bias[col];
                if (mode == 0) {
                    ((float*)out)[(size_t)row * N + col] = v;
                } else {
                    int b = row >> 11, t = row & 2047;
                    int h = col >> 6,  d = col & 63;
                    ((unsigned short*)out)[((size_t)(b * Hh + h) * DHd + d) * Tt + t] = f2bf(v);
                }
            }
}

// =====================================================================
// RoPE + pack f32 [B*T, D] -> bf16 [B,H,T,DH] for Q and K
// =====================================================================
__global__ __launch_bounds__(256) void rope_pack_kernel(
    const float* __restrict__ Qf, const float* __restrict__ Kf,
    unsigned short* __restrict__ Qb, unsigned short* __restrict__ Kb)
{
    int gid = blockIdx.x * 256 + threadIdx.x;      // B*T*H*DH threads
    int d = gid & 63;
    int h = (gid >> 6) & 15;
    int t = (gid >> 10) & 2047;
    int b = gid >> 21;
    size_t src = (size_t)(b * Tt + t) * Dm + h * DHd + d;
    size_t par = src + ((d < 32) ? 32 : -32);
    float sign = (d < 32) ? -1.f : 1.f;
    float ang = (float)t * __expf(-(float)(d >> 1) * (9.210340371976184f / 32.f));
    float s = __sinf(ang), c = __cosf(ang);
    size_t dst = ((size_t)(b * Hh + h) * Tt + t) * DHd + d;
    float q = Qf[src], qp = Qf[par];
    Qb[dst] = f2bf(q * c + sign * qp * s);
    float k = Kf[src], kp = Kf[par];
    Kb[dst] = f2bf(k * c + sign * kp * s);
}

// =====================================================================
// TDM helper: 2-D tile load Global -> LDS with LDS row padding.
//   tile_x elements contiguous per row, tile_y rows, row stride (elements),
//   data_size = 2 bytes. pad_int/pad_amt encode D# padding codes.
// =====================================================================
__device__ __forceinline__ void tdm_load_2d(
    unsigned lds_addr, unsigned long long ga,
    int tile_x, int tile_y, int tensor_x, int tensor_y,
    int row_stride, int pad_int, int pad_amt)
{
    u32x4 g0;
    g0[0] = 1u;                                   // count=1, user descriptor
    g0[1] = lds_addr;
    g0[2] = (unsigned)ga;
    g0[3] = ((unsigned)(ga >> 32) & 0x01FFFFFFu) | 0x80000000u;   // type=2
    i32x8 g1;
    g1[0] = (int)((1u << 16) | (1u << 20) |       // data_size=2B, pad_enable
                  ((unsigned)pad_int << 22) | ((unsigned)pad_amt << 25));
    g1[1] = (int)(((unsigned)tensor_x & 0xFFFFu) << 16);          // tensor_dim0 lo
    g1[2] = (int)((((unsigned)tensor_x >> 16) & 0xFFFFu) |
                  (((unsigned)tensor_y & 0xFFFFu) << 16));        // td0 hi | td1 lo
    g1[3] = (int)((((unsigned)tensor_y >> 16) & 0xFFFFu) |
                  (((unsigned)tile_x) << 16));                    // td1 hi | tile_dim0
    g1[4] = tile_y;                                               // tile_dim1 (tile_dim2=0)
    g1[5] = row_stride;                                           // tensor_dim0_stride lo
    g1[6] = 0;
    g1[7] = 0;
    i32x4 gz4 = {0, 0, 0, 0};
    i32x8 gz8 = {0, 0, 0, 0, 0, 0, 0, 0};
    __builtin_amdgcn_tensor_load_to_lds(g0, g1, gz4, gz4, gz8, 0);
}

// =====================================================================
// Flash attention: grid = (T/64, B*H), 128 threads (4 waves x 16 queries)
// K/V tiles staged in LDS by the Tensor Data Mover.
// =====================================================================
__global__ __launch_bounds__(128) void attn_kernel(
    const unsigned short* __restrict__ Qb,   // [BH, T, 64] bf16
    const unsigned short* __restrict__ Kb,   // [BH, T, 64] bf16
    const unsigned short* __restrict__ Vt,   // [BH, 64, T] bf16
    float* __restrict__ Oattn)               // [B*T, 1024] f32
{
    __shared__ __align__(16) unsigned short sK[32][72];      // [key][dh]  (64+8 pad)
    __shared__ __align__(16) unsigned short sV[64][40];      // [dh][key]  (32+8 pad)
    __shared__ __align__(16) unsigned short sP[4][16][40];   // per-wave P tile

    const int lane = threadIdx.x & 31;
    const int wave = threadIdx.x >> 5;
    const int bh = blockIdx.y;
    const int b  = bh >> 4;
    const int h  = bh & 15;
    const int qblk = blockIdx.x * 64;
    const int qw   = qblk + wave * 16;
    const int l16   = lane & 15;
    const int half8 = (lane >> 4) * 8;
    const int halfo = (lane >> 4) * 8;

    // Q A-fragments (dh 0..31 and 32..63)
    const unsigned short* qrow = Qb + ((size_t)bh * Tt + (qw + l16)) * DHd;
    Frag a0, a1;
    a0.u[0] = *(const u32x4*)(qrow + half8);
    a0.u[1] = *(const u32x4*)(qrow + half8 + 16);
    a1.u[0] = *(const u32x4*)(qrow + 32 + half8);
    a1.u[1] = *(const u32x4*)(qrow + 32 + half8 + 16);

    f32x8 o0 = {}, o1 = {}, o2 = {}, o3 = {};
    float mrow[8], lrow[8];
    for (int i = 0; i < 8; ++i) { mrow[i] = -1e30f; lrow[i] = 0.f; }

    const unsigned ldsK = (unsigned)(size_t)&sK[0][0];
    const unsigned ldsV = (unsigned)(size_t)&sV[0][0];

    const int nkt   = qblk / 32 + 2;   // key tiles staged by the block
    const int ktend = qw / 32 + 1;     // key tiles this wave needs

    for (int kt = 0; kt < nkt; ++kt) {
        const int kbase = kt * 32;
        if (wave == 0) {
            // K tile: 32 key-rows of 64 contiguous dh; LDS rows padded 64->72
            unsigned long long gaK = (unsigned long long)(size_t)
                (Kb + ((size_t)bh * Tt + kbase) * DHd);
            tdm_load_2d(ldsK, gaK, /*tile_x=*/64, /*tile_y=*/32,
                        /*tensor_x=*/64, /*tensor_y=*/Tt,
                        /*row_stride=*/64, /*pad_int: 32DW*/4, /*pad_amt: 4DW*/3);
            // V tile: 64 dh-rows of 32 contiguous keys; LDS rows padded 32->40
            unsigned long long gaV = (unsigned long long)(size_t)
                (Vt + (size_t)bh * DHd * Tt + kbase);
            tdm_load_2d(ldsV, gaV, /*tile_x=*/32, /*tile_y=*/64,
                        /*tensor_x=*/Tt, /*tensor_y=*/64,
                        /*row_stride=*/Tt, /*pad_int: 16DW*/3, /*pad_amt: 4DW*/3);
            __builtin_amdgcn_s_wait_tensorcnt(0);
        }
        __syncthreads();

        if (kt < ktend) {
            // S = Q K^T (two 16-key column tiles, two 32-dh k-steps)
            Frag bk00, bk01, bk10, bk11;
            {
                int key0 = l16, key1 = 16 + l16;
                bk00.u[0] = *(const u32x4*)&sK[key0][half8];
                bk00.u[1] = *(const u32x4*)&sK[key0][half8 + 16];
                bk01.u[0] = *(const u32x4*)&sK[key0][32 + half8];
                bk01.u[1] = *(const u32x4*)&sK[key0][32 + half8 + 16];
                bk10.u[0] = *(const u32x4*)&sK[key1][half8];
                bk10.u[1] = *(const u32x4*)&sK[key1][half8 + 16];
                bk11.u[0] = *(const u32x4*)&sK[key1][32 + half8];
                bk11.u[1] = *(const u32x4*)&sK[key1][32 + half8 + 16];
            }
            f32x8 s0 = {}, s1 = {};
            s0 = wmma_bf16(a0.v, bk00.v, s0);
            s0 = wmma_bf16(a1.v, bk01.v, s0);
            s1 = wmma_bf16(a0.v, bk10.v, s1);
            s1 = wmma_bf16(a1.v, bk11.v, s1);

            // scale + causal mask
            for (int i = 0; i < 8; ++i) {
                int q = qw + halfo + i;
                float v0 = s0[i] * 0.125f;   // 1/sqrt(64)
                float v1 = s1[i] * 0.125f;
                if (kbase + l16 > q)      v0 = -1e30f;
                if (kbase + 16 + l16 > q) v1 = -1e30f;
                s0[i] = v0; s1[i] = v1;
            }

            // online softmax (row reductions inside each 16-lane half)
            float alpha[8];
            for (int i = 0; i < 8; ++i) {
                float mx = fmaxf(s0[i], s1[i]);
                for (int dd = 8; dd >= 1; dd >>= 1)
                    mx = fmaxf(mx, __shfl_xor(mx, dd, 32));
                float mn = fmaxf(mrow[i], mx);
                alpha[i] = __expf(mrow[i] - mn);
                mrow[i]  = mn;
                float p0 = __expf(s0[i] - mn);
                float p1 = __expf(s1[i] - mn);
                s0[i] = p0; s1[i] = p1;
                float rs = p0 + p1;
                for (int dd = 8; dd >= 1; dd >>= 1)
                    rs += __shfl_xor(rs, dd, 32);
                lrow[i] = lrow[i] * alpha[i] + rs;
                o0[i] *= alpha[i]; o1[i] *= alpha[i];
                o2[i] *= alpha[i]; o3[i] *= alpha[i];
            }

            // P: C-layout -> per-wave LDS -> reload as A-fragment
            for (int i = 0; i < 8; ++i) {
                int r = halfo + i;
                sP[wave][r][l16]      = f2bf(s0[i]);
                sP[wave][r][16 + l16] = f2bf(s1[i]);
            }
            asm volatile("s_wait_dscnt 0" ::: "memory");
            __builtin_amdgcn_wave_barrier();

            Frag pf;
            pf.u[0] = *(const u32x4*)&sP[wave][l16][half8];
            pf.u[1] = *(const u32x4*)&sP[wave][l16][half8 + 16];

            Frag bv0, bv1, bv2, bv3;
            bv0.u[0] = *(const u32x4*)&sV[l16][half8];
            bv0.u[1] = *(const u32x4*)&sV[l16][half8 + 16];
            bv1.u[0] = *(const u32x4*)&sV[16 + l16][half8];
            bv1.u[1] = *(const u32x4*)&sV[16 + l16][half8 + 16];
            bv2.u[0] = *(const u32x4*)&sV[32 + l16][half8];
            bv2.u[1] = *(const u32x4*)&sV[32 + l16][half8 + 16];
            bv3.u[0] = *(const u32x4*)&sV[48 + l16][half8];
            bv3.u[1] = *(const u32x4*)&sV[48 + l16][half8 + 16];

            o0 = wmma_bf16(pf.v, bv0.v, o0);
            o1 = wmma_bf16(pf.v, bv1.v, o1);
            o2 = wmma_bf16(pf.v, bv2.v, o2);
            o3 = wmma_bf16(pf.v, bv3.v, o3);
        }
        __syncthreads();
    }

    // epilogue: Oattn[b, q, h, dh] f32
    for (int i = 0; i < 8; ++i) {
        int q = qw + halfo + i;
        float inv = 1.f / lrow[i];
        float* orow = Oattn + ((size_t)(b * Tt + q)) * Dm + h * DHd;
        orow[l16]      = o0[i] * inv;
        orow[16 + l16] = o1[i] * inv;
        orow[32 + l16] = o2[i] * inv;
        orow[48 + l16] = o3[i] * inv;
    }
}

// =====================================================================
extern "C" void kernel_launch(void* const* d_in, const int* in_sizes, int n_in,
                              void* d_out, int out_size, void* d_ws, size_t ws_size,
                              hipStream_t stream) {
    const float* x  = (const float*)d_in[0];
    // d_in[1] = mask (causal, implicit)
    const float* wq = (const float*)d_in[2];
    const float* bq = (const float*)d_in[3];
    const float* wk = (const float*)d_in[4];
    const float* bk = (const float*)d_in[5];
    const float* wv = (const float*)d_in[6];
    const float* bv = (const float*)d_in[7];
    const float* wo = (const float*)d_in[8];
    const float* bo = (const float*)d_in[9];
    float* out = (float*)d_out;

    char* ws = (char*)d_ws;
    const size_t MB = 1024 * 1024;
    float*          Qf    = (float*)(ws + 0);                  // 16 MB
    float*          Kf    = (float*)(ws + 16 * MB);            // 16 MB
    float*          Oattn = (float*)(ws + 32 * MB);            // 16 MB
    unsigned short* Qb    = (unsigned short*)(ws + 48 * MB);   // 8 MB
    unsigned short* Kb    = (unsigned short*)(ws + 56 * MB);   // 8 MB
    unsigned short* Vt    = (unsigned short*)(ws + 64 * MB);   // 8 MB
    unsigned short* xb    = (unsigned short*)(ws + 72 * MB);   // 8 MB
    unsigned short* Ob    = (unsigned short*)(ws + 80 * MB);   // 8 MB
    unsigned short* Wtq   = (unsigned short*)(ws + 88 * MB);   // 2 MB
    unsigned short* Wtk   = (unsigned short*)(ws + 90 * MB);   // 2 MB
    unsigned short* Wtv   = (unsigned short*)(ws + 92 * MB);   // 2 MB
    unsigned short* Wto   = (unsigned short*)(ws + 94 * MB);   // 2 MB

    dim3 tgrid(Dm / 32, Dm / 32);            // (32, 32)
    convert_bf16_kernel<<<(BT * Dm) / 1024, 256, 0, stream>>>(x, xb);
    transpose_w_kernel<<<tgrid, 256, 0, stream>>>(wq, Wtq);
    transpose_w_kernel<<<tgrid, 256, 0, stream>>>(wk, Wtk);
    transpose_w_kernel<<<tgrid, 256, 0, stream>>>(wv, Wtv);
    transpose_w_kernel<<<tgrid, 256, 0, stream>>>(wo, Wto);

    dim3 ggrid(Dm / 64, BT / 64);            // (16, 64)
    gemm_bf16_kernel<<<ggrid, 128, 0, stream>>>(xb, Wtq, bq, Qf, BT, Dm, Dm, 0);
    gemm_bf16_kernel<<<ggrid, 128, 0, stream>>>(xb, Wtk, bk, Kf, BT, Dm, Dm, 0);
    gemm_bf16_kernel<<<ggrid, 128, 0, stream>>>(xb, Wtv, bv, Vt, BT, Dm, Dm, 1);

    rope_pack_kernel<<<(BT * Dm) / 256, 256, 0, stream>>>(Qf, Kf, Qb, Kb);

    dim3 agrid(Tt / 64, 2 * Hh);             // (32, 32)
    attn_kernel<<<agrid, 128, 0, stream>>>(Qb, Kb, Vt, Oattn);

    convert_bf16_kernel<<<(BT * Dm) / 1024, 256, 0, stream>>>(Oattn, Ob);
    gemm_bf16_kernel<<<ggrid, 128, 0, stream>>>(Ob, Wto, bo, out, BT, Dm, Dm, 0);
}
